// RWKV6Model_42743514530299
// MI455X (gfx1250) — matrix operand order repaired and checked by reading, not compiled, not run on hardware
//
#include <hip/hip_runtime.h>
#include <hip/hip_bf16.h>
#include <math.h>

// ---------------------------------------------------------------------------
// RWKV-6 forward for MI455X (gfx1250, wave32, WMMA, async global->LDS).
//   B=2, T=1024, D=512, H=4, Dh=128, F=2048, L=8, V=32000
// All GEMMs: bf16 operands (weights pre-converted+transposed to [N,K] bf16),
// fp32 WMMA accumulation, double-buffered LDS tiles staged with
// GLOBAL_LOAD_ASYNC_TO_LDS_B128 (ASYNCcnt path).
// ---------------------------------------------------------------------------

#define BV   32000
#define BD   512
#define BL   8
#define BH   4
#define BDh  128
#define BF   2048
#define BB   2
#define BT   1024
#define BN   (BB*BT)       // 2048 token rows

typedef unsigned short ushort_t;
typedef __attribute__((ext_vector_type(16))) __bf16        v16bf;
typedef __attribute__((ext_vector_type(8)))  float         v8f;
typedef __attribute__((ext_vector_type(4)))  unsigned int  v4u;

union FragU { v16bf bf; v4u u4[2]; };

__device__ __forceinline__ ushort_t f2bf(float f) {
  unsigned u = __builtin_bit_cast(unsigned, f);
  unsigned r = u + 0x7FFFu + ((u >> 16) & 1u);   // round-to-nearest-even
  return (ushort_t)(r >> 16);
}

// ---- gfx1250 async global->LDS copy (ASYNCcnt path), with fallback ---------
#if defined(__has_builtin)
#if __has_builtin(__builtin_amdgcn_global_load_async_to_lds_b128) && \
    __has_builtin(__builtin_amdgcn_s_wait_asynccnt)
#define USE_ASYNC_LDS 1
#endif
#endif
#ifndef USE_ASYNC_LDS
#define USE_ASYNC_LDS 0
#endif

#define GAS __attribute__((address_space(1)))
#define LAS __attribute__((address_space(3)))
typedef int v4i_vs __attribute__((vector_size(16)));   // matches builtin pointee

__device__ __forceinline__ void copy_b128_g2l(const ushort_t* g, ushort_t* l) {
#if USE_ASYNC_LDS
  GAS v4i_vs* gp = (GAS v4i_vs*)(GAS void*)(ushort_t*)g;
  LAS v4i_vs* lp = (LAS v4i_vs*)(LAS void*)l;
  __builtin_amdgcn_global_load_async_to_lds_b128(gp, lp, 0, 0);
#else
  *(v4u*)l = *(const v4u*)g;
#endif
}
__device__ __forceinline__ void copy_wait() {
#if USE_ASYNC_LDS
  __builtin_amdgcn_s_wait_asynccnt(0);
#endif
}

// ---------------------------------------------------------------------------
// GEMM: D[M,N] = A[M,K](bf16) x Bt[N,K](bf16)  via v_wmma_f32_16x16x32_bf16
// MODE 0: outF = acc
// MODE 1: outF = Cres + acc                (residual add)
// MODE 2: outF = exp(-exp(acc + bias[n]))  (RWKV decay w)
// MODE 3: outBF = bf16(gelu_exact(acc))    (FFN mid)
// Block tile BM x 128, 256 threads = 8 waves, double-buffered LDS.
// ---------------------------------------------------------------------------
template<int MODE, int BM>
__global__ __launch_bounds__(256)
void gemm_bf16_kernel(const ushort_t* __restrict__ A, const ushort_t* __restrict__ Bt,
                      const float* __restrict__ Cres, const float* __restrict__ bias,
                      float* __restrict__ outF, ushort_t* __restrict__ outBF,
                      int M, int N, int K)
{
  constexpr int WR  = BM / 32;        // waves along M
  constexpr int WC  = 8 / WR;         // waves along N
  constexpr int WN  = 128 / WC;       // cols per wave
  constexpr int NI  = WN / 16;        // 16-wide n-frags per wave
  constexpr int ACH = (BM * 4) / 256; // A b128-chunks per thread

  __shared__ __align__(16) ushort_t As[2][BM * 32];
  __shared__ __align__(16) ushort_t Bs[2][128 * 32];

  const int tid  = threadIdx.x;
  const int lane = tid & 31;
  const int wid  = tid >> 5;
  const int bm0  = blockIdx.y * BM;
  const int bn0  = blockIdx.x * 128;
  const int wm0  = (wid % WR) * 32;
  const int wn0  = (wid / WR) * WN;
  const int hi   = (lane >> 4) & 1;
  const int l15  = lane & 15;

  v8f acc[2][NI] = {};

  auto stage = [&](int it, int buf) {
    const int kb = it * 32;
    #pragma unroll
    for (int i = 0; i < ACH; i++) {                 // A: BM x 32 halves
      int c = tid + i * 256;
      int row = c >> 2, koff = (c & 3) * 8;
      copy_b128_g2l(A + (size_t)(bm0 + row) * K + kb + koff,
                    &As[buf][row * 32 + koff]);
    }
    #pragma unroll
    for (int i = 0; i < 2; i++) {                   // B: 128 x 32 halves
      int c = tid + i * 256;
      int n = c >> 2, koff = (c & 3) * 8;
      copy_b128_g2l(Bt + (size_t)(bn0 + n) * K + kb + koff,
                    &Bs[buf][n * 32 + koff]);
    }
  };

  const int nIt = K >> 5;
  stage(0, 0);
  copy_wait();
  __syncthreads();

  for (int it = 0; it < nIt; it++) {
    const int cur = it & 1, nxt = cur ^ 1;
    if (it + 1 < nIt) stage(it + 1, nxt);

    // fragments per ISA 16-bit layouts
    FragU a[2], b[NI];
    #pragma unroll
    for (int mi = 0; mi < 2; mi++) {
      const ushort_t* p = &As[cur][(wm0 + mi * 16 + l15) * 32 + hi * 8];
      a[mi].u4[0] = *(const v4u*)(p);          // K = hi*8 .. +7
      a[mi].u4[1] = *(const v4u*)(p + 16);     // K = 16+hi*8 .. +7
    }
    #pragma unroll
    for (int ni = 0; ni < NI; ni++) {
      const ushort_t* p = &Bs[cur][(wn0 + ni * 16 + l15) * 32 + hi * 8];
      b[ni].u4[0] = *(const v4u*)(p);          // K = hi*8 .. +7
      b[ni].u4[1] = *(const v4u*)(p + 16);     // K = 16+hi*8 .. +7
    }
    #pragma unroll
    for (int mi = 0; mi < 2; mi++)
      #pragma unroll
      for (int ni = 0; ni < NI; ni++)
        acc[mi][ni] = __builtin_amdgcn_wmma_f32_16x16x32_bf16(
            false, a[mi].bf, false, b[ni].bf, (short)0, acc[mi][ni], false, false);

    if (it + 1 < nIt) copy_wait();
    __syncthreads();
  }

  // epilogue: C layout (VGPR r -> M = r + 8*hi, N = lane&15)
  #pragma unroll
  for (int mi = 0; mi < 2; mi++)
    #pragma unroll
    for (int ni = 0; ni < NI; ni++) {
      int col = bn0 + wn0 + ni * 16 + l15;
      #pragma unroll
      for (int r = 0; r < 8; r++) {
        int row = bm0 + wm0 + mi * 16 + hi * 8 + r;
        size_t idx = (size_t)row * N + col;
        float v = acc[mi][ni][r];
        if (MODE == 0) {
          outF[idx] = v;
        } else if (MODE == 1) {
          outF[idx] = Cres[idx] + v;
        } else if (MODE == 2) {
          float z = v + bias[col];
          outF[idx] = __expf(-__expf(z));
        } else { // MODE 3: exact gelu -> bf16
          float g = 0.5f * v * (1.0f + erff(v * 0.70710678118654752f));
          outBF[idx] = f2bf(g);
        }
      }
    }
}

// ---------------------------------------------------------------------------
// Weight prep: fp32 [K,N] -> bf16 [N,K] (LDS-tiled transpose, coalesced both ways)
// ---------------------------------------------------------------------------
__global__ __launch_bounds__(256)
void wconv_t_kernel(const float* __restrict__ in, ushort_t* __restrict__ out,
                    int K, int N)
{
  __shared__ float tile[32][33];
  int kb = blockIdx.y * 32, nb = blockIdx.x * 32;
  int tx = threadIdx.x & 31, ty = threadIdx.x >> 5;  // 32 x 8
  #pragma unroll
  for (int i = 0; i < 4; i++)
    tile[ty + i * 8][tx] = in[(size_t)(kb + ty + i * 8) * N + nb + tx];
  __syncthreads();
  #pragma unroll
  for (int i = 0; i < 4; i++)
    out[(size_t)(nb + ty + i * 8) * K + kb + tx] = f2bf(tile[tx][ty + i * 8]);
}

// fp32 -> bf16 elementwise (x4 vectorized), for the embedding matrix
__global__ void cvt_bf16x4_kernel(const float* __restrict__ in,
                                  ushort_t* __restrict__ out, long n4)
{
  long i = (long)blockIdx.x * blockDim.x + threadIdx.x;
  if (i >= n4) return;
  float4 f = ((const float4*)in)[i];
  ushort4 u;
  u.x = f2bf(f.x); u.y = f2bf(f.y); u.z = f2bf(f.z); u.w = f2bf(f.w);
  ((ushort4*)out)[i] = u;
}

// ---------------------------------------------------------------------------
// Embedding gather: X[n,:] = embed[tokens[n],:]
// ---------------------------------------------------------------------------
__global__ void embed_kernel(const int* __restrict__ tok, const float* __restrict__ emb,
                             float* __restrict__ X)
{
  int n = blockIdx.x;
  int t = tok[n];
  for (int d = threadIdx.x; d < BD; d += blockDim.x)
    X[(size_t)n * BD + d] = emb[(size_t)t * BD + d];
}

// ---------------------------------------------------------------------------
// rmsnorm stats: rstd[n] = rsqrt(mean(x^2) + 1e-6)
// ---------------------------------------------------------------------------
__global__ __launch_bounds__(256)
void rms_stats_kernel(const float* __restrict__ X, float* __restrict__ rstd)
{
  __shared__ float red[256];
  int n = blockIdx.x;
  const float* x = X + (size_t)n * BD;
  float s = 0.f;
  for (int d = threadIdx.x; d < BD; d += 256) { float v = x[d]; s += v * v; }
  red[threadIdx.x] = s;
  __syncthreads();
  for (int off = 128; off > 0; off >>= 1) {
    if (threadIdx.x < off) red[threadIdx.x] += red[threadIdx.x + off];
    __syncthreads();
  }
  if (threadIdx.x == 0) rstd[n] = rsqrtf(red[0] * (1.0f / BD) + 1e-6f);
}

// ---------------------------------------------------------------------------
// hn = rmsnorm(x); token-shift mix: out[j] = bf16(hn + (hn_prev - hn)*mix[j])
// ---------------------------------------------------------------------------
__global__ __launch_bounds__(256)
void mix_kernel(const float* __restrict__ X, const float* __restrict__ rstd,
                const float* __restrict__ ln1, const float* __restrict__ mix,
                ushort_t* __restrict__ out)
{
  int n = blockIdx.x;
  int t = n % BT;
  float rs = rstd[n];
  float rp = (t > 0) ? rstd[n - 1] : 0.f;
  for (int d = threadIdx.x; d < BD; d += 256) {
    float w  = ln1[d];
    float hn = X[(size_t)n * BD + d] * rs * w;
    float hp = (t > 0) ? X[(size_t)(n - 1) * BD + d] * rp * w : 0.f;
    float xx = hp - hn;
    #pragma unroll
    for (int j = 0; j < 5; j++)
      out[((size_t)j * BN + n) * BD + d] = f2bf(hn + xx * mix[j * BD + d]);
  }
}

// out = bf16(rmsnorm(x, w))
__global__ __launch_bounds__(256)
void rms_apply_kernel(const float* __restrict__ X, const float* __restrict__ rstd,
                      const float* __restrict__ w, ushort_t* __restrict__ out)
{
  int n = blockIdx.x;
  float rs = rstd[n];
  for (int d = threadIdx.x; d < BD; d += 256)
    out[(size_t)n * BD + d] = f2bf(X[(size_t)n * BD + d] * rs * w[d]);
}

// ---------------------------------------------------------------------------
// WKV linear recurrence. One block per (b,h); 128 threads (thread = v index).
// State S[128][128] in dynamic LDS (64 KB << 320 KB/WGP).
// ---------------------------------------------------------------------------
__global__ __launch_bounds__(128)
void wkv_kernel(const float* __restrict__ R, const float* __restrict__ Kb_,
                const float* __restrict__ V, const float* __restrict__ W,
                const float* __restrict__ U, float* __restrict__ Y)
{
  extern __shared__ float S[];                      // BDh*BDh
  __shared__ float rb[BDh], kb[BDh], wb[BDh], vb[BDh], ub[BDh], red[BDh];
  int bh = blockIdx.x;
  int b = bh / BH, h = bh % BH;
  int tid = threadIdx.x;

  ub[tid] = U[h * BDh + tid];
  for (int i = tid; i < BDh * BDh; i += 128) S[i] = 0.f;
  __syncthreads();

  for (int t = 0; t < BT; t++) {
    size_t base = ((size_t)(b * BT + t)) * BD + h * BDh;
    rb[tid] = R[base + tid];
    kb[tid] = Kb_[base + tid];
    wb[tid] = W[base + tid];
    vb[tid] = V[base + tid];
    __syncthreads();
    red[tid] = rb[tid] * ub[tid] * kb[tid];
    __syncthreads();
    for (int off = 64; off > 0; off >>= 1) {
      if (tid < off) red[tid] += red[tid + off];
      __syncthreads();
    }
    float vj = vb[tid];
    float y  = red[0] * vj;
    #pragma unroll 4
    for (int k = 0; k < BDh; k++) {
      float s = S[k * BDh + tid];
      y += rb[k] * s;
      S[k * BDh + tid] = wb[k] * s + kb[k] * vj;
    }
    Y[base + tid] = y;
    __syncthreads();
  }
}

// ---------------------------------------------------------------------------
// Per-head GroupNorm + silu gate: out = bf16( GN(y)*gn_w + gn_b ) * silu(g)
// ---------------------------------------------------------------------------
__global__ __launch_bounds__(256)
void gn_gate_kernel(const float* __restrict__ Yb, const float* __restrict__ G,
                    const float* __restrict__ gnw, const float* __restrict__ gnb,
                    ushort_t* __restrict__ out)
{
  __shared__ float buf[BD];
  __shared__ float r1[256], r2[256];
  __shared__ float mu[BH], rs[BH];
  int n = blockIdx.x, tid = threadIdx.x;
  const float* y = Yb + (size_t)n * BD;
  buf[tid] = y[tid];
  buf[tid + 256] = y[tid + 256];
  __syncthreads();

  int grp = tid >> 6, l = tid & 63;                 // 4 heads x 64 threads
  float a = buf[grp * 128 + l], b = buf[grp * 128 + l + 64];
  r1[tid] = a + b;
  r2[tid] = a * a + b * b;
  __syncthreads();
  for (int off = 32; off > 0; off >>= 1) {
    if (l < off) { r1[tid] += r1[tid + off]; r2[tid] += r2[tid + off]; }
    __syncthreads();
  }
  if (l == 0) {
    float m = r1[grp * 64] * (1.0f / BDh);
    mu[grp] = m;
    rs[grp] = rsqrtf(r2[grp * 64] * (1.0f / BDh) - m * m + 1e-5f);
  }
  __syncthreads();

  #pragma unroll
  for (int i = 0; i < 2; i++) {
    int d = tid + i * 256;
    int h = d >> 7;
    float v = (buf[d] - mu[h]) * rs[h] * gnw[d] + gnb[d];
    float g = G[(size_t)n * BD + d];
    g = g / (1.0f + __expf(-g));                    // silu
    out[(size_t)n * BD + d] = f2bf(v * g);
  }
}

// ---------------------------------------------------------------------------
// Host-side orchestration
// ---------------------------------------------------------------------------
extern "C" void kernel_launch(void* const* d_in, const int* in_sizes, int n_in,
                              void* d_out, int out_size, void* d_ws, size_t ws_size,
                              hipStream_t stream)
{
  (void)in_sizes; (void)n_in; (void)out_size; (void)ws_size;

  const int*   tokens = (const int*)  d_in[0];
  const float* embed  = (const float*)d_in[1];
  const float* ln1_w  = (const float*)d_in[2];
  const float* ln2_w  = (const float*)d_in[3];
  const float* lnout  = (const float*)d_in[4];
  const float* mix    = (const float*)d_in[5];
  const float* Wr     = (const float*)d_in[6];
  const float* Wk     = (const float*)d_in[7];
  const float* Wv     = (const float*)d_in[8];
  const float* Wg     = (const float*)d_in[9];
  const float* Ww     = (const float*)d_in[10];
  const float* w0     = (const float*)d_in[11];
  const float* Wo     = (const float*)d_in[12];
  const float* u      = (const float*)d_in[13];
  const float* gn_w   = (const float*)d_in[14];
  const float* gn_b   = (const float*)d_in[15];
  const float* W1     = (const float*)d_in[16];
  const float* W2     = (const float*)d_in[17];

  // ---- workspace carve-up (~86 MB) ----
  char* wp = (char*)d_ws;
  auto carve = [&](size_t bytes) -> void* {
    void* p = (void*)wp;
    wp += (bytes + 255) & ~(size_t)255;
    return p;
  };
  float*    X    = (float*)   carve((size_t)BN * BD * 4);       // residual stream
  float*    rstd = (float*)   carve((size_t)BN * 4);
  ushort_t* TBF  = (ushort_t*)carve((size_t)5 * BN * BD * 2);   // xr..xg bf16
  float*    Rb   = (float*)   carve((size_t)BN * BD * 4);
  float*    Kb   = (float*)   carve((size_t)BN * BD * 4);
  float*    Vb   = (float*)   carve((size_t)BN * BD * 4);
  float*    Wb   = (float*)   carve((size_t)BN * BD * 4);
  float*    Gb   = (float*)   carve((size_t)BN * BD * 4);
  float*    Yb   = (float*)   carve((size_t)BN * BD * 4);
  ushort_t* MID  = (ushort_t*)carve((size_t)BN * BF * 2);       // FFN mid bf16
  ushort_t* EMB16= (ushort_t*)carve((size_t)BV * BD * 2);       // embed bf16 [V,D]
  const size_t DD = (size_t)BD * BD;
  ushort_t* WT   = (ushort_t*)carve((size_t)(6 * DD + 2 * (size_t)BD * BF) * 2);
  ushort_t* WrT = WT;                 // [D,D]^T bf16
  ushort_t* WkT = WrT + DD;
  ushort_t* WvT = WkT + DD;
  ushort_t* WgT = WvT + DD;
  ushort_t* WwT = WgT + DD;
  ushort_t* WoT = WwT + DD;
  ushort_t* W1T = WoT + DD;           // [F,D] bf16
  ushort_t* W2T = W1T + (size_t)BF * BD;  // [D,F] bf16
  ushort_t* T0  = TBF;                // slice-0 reuse

  const dim3 blk(256);
  const dim3 gDD(BD / 128, BN / 64);      // N=512  GEMMs, BM=64  -> 128 blocks
  const dim3 gDF(BF / 128, BN / 128);     // N=2048 GEMM,  BM=128 -> 256 blocks
  const dim3 gDV(BV / 128, BN / 128);     // logits GEMM          -> 4000 blocks
  const dim3 gTT(BD / 32, BD / 32);       // D x D transpose-convert

  // embedding: gather rows + one-time bf16 copy for the logits GEMM
  embed_kernel<<<BN, blk, 0, stream>>>(tokens, embed, X);
  {
    long n4 = (long)BV * BD / 4;
    cvt_bf16x4_kernel<<<dim3((unsigned)((n4 + 255) / 256)), blk, 0, stream>>>(embed, EMB16, n4);
  }

  for (int i = 0; i < BL; i++) {
    // per-layer weight prep: fp32 [K,N] -> bf16 [N,K]
    wconv_t_kernel<<<gTT, blk, 0, stream>>>(Wr + (size_t)i*DD, WrT, BD, BD);
    wconv_t_kernel<<<gTT, blk, 0, stream>>>(Wk + (size_t)i*DD, WkT, BD, BD);
    wconv_t_kernel<<<gTT, blk, 0, stream>>>(Wv + (size_t)i*DD, WvT, BD, BD);
    wconv_t_kernel<<<gTT, blk, 0, stream>>>(Wg + (size_t)i*DD, WgT, BD, BD);
    wconv_t_kernel<<<gTT, blk, 0, stream>>>(Ww + (size_t)i*DD, WwT, BD, BD);
    wconv_t_kernel<<<gTT, blk, 0, stream>>>(Wo + (size_t)i*DD, WoT, BD, BD);
    wconv_t_kernel<<<dim3(BF/32, BD/32), blk, 0, stream>>>(W1 + (size_t)i*BD*BF, W1T, BD, BF);
    wconv_t_kernel<<<dim3(BD/32, BF/32), blk, 0, stream>>>(W2 + (size_t)i*BF*BD, W2T, BF, BD);

    // rmsnorm + token-shift mixing -> 5 bf16 activation mats
    rms_stats_kernel<<<BN, blk, 0, stream>>>(X, rstd);
    mix_kernel<<<BN, blk, 0, stream>>>(X, rstd, ln1_w + (size_t)i*BD, mix + (size_t)i*5*BD, TBF);

    const ushort_t* xr = TBF + (size_t)0 * BN * BD;
    const ushort_t* xk = TBF + (size_t)1 * BN * BD;
    const ushort_t* xv = TBF + (size_t)2 * BN * BD;
    const ushort_t* xw = TBF + (size_t)3 * BN * BD;
    const ushort_t* xg = TBF + (size_t)4 * BN * BD;

    gemm_bf16_kernel<0,64><<<gDD, blk, 0, stream>>>(xr, WrT, nullptr, nullptr, Rb, nullptr, BN, BD, BD);
    gemm_bf16_kernel<0,64><<<gDD, blk, 0, stream>>>(xk, WkT, nullptr, nullptr, Kb, nullptr, BN, BD, BD);
    gemm_bf16_kernel<0,64><<<gDD, blk, 0, stream>>>(xv, WvT, nullptr, nullptr, Vb, nullptr, BN, BD, BD);
    gemm_bf16_kernel<2,64><<<gDD, blk, 0, stream>>>(xw, WwT, nullptr, w0 + (size_t)i*BD, Wb, nullptr, BN, BD, BD);
    gemm_bf16_kernel<0,64><<<gDD, blk, 0, stream>>>(xg, WgT, nullptr, nullptr, Gb, nullptr, BN, BD, BD);

    // sequential WKV scan: one block per (b,h), 64 KB LDS state
    wkv_kernel<<<BB * BH, dim3(128), BDh * BDh * sizeof(float), stream>>>(
        Rb, Kb, Vb, Wb, u + (size_t)i * BH * BDh, Yb);

    // GroupNorm + silu gate -> bf16 A for Wo
    gn_gate_kernel<<<BN, blk, 0, stream>>>(Yb, Gb, gn_w + (size_t)i*BD, gn_b + (size_t)i*BD, T0);

    // x += (y*g) @ Wo   (residual fused)
    gemm_bf16_kernel<1,64><<<gDD, blk, 0, stream>>>(T0, WoT, X, nullptr, X, nullptr, BN, BD, BD);

    // FFN: x += gelu(rmsnorm(x) @ W1) @ W2
    rms_stats_kernel<<<BN, blk, 0, stream>>>(X, rstd);
    rms_apply_kernel<<<BN, blk, 0, stream>>>(X, rstd, ln2_w + (size_t)i*BD, T0);
    gemm_bf16_kernel<3,128><<<gDF, blk, 0, stream>>>(T0, W1T, nullptr, nullptr, nullptr, MID, BN, BF, BD);
    gemm_bf16_kernel<1,64><<<gDD, blk, 0, stream>>>(MID, W2T, X, nullptr, X, nullptr, BN, BD, BF);
  }

  // final norm + tied-embedding logits: out = rmsnorm(x) @ embed^T
  rms_stats_kernel<<<BN, blk, 0, stream>>>(X, rstd);
  rms_apply_kernel<<<BN, blk, 0, stream>>>(X, rstd, lnout, T0);
  gemm_bf16_kernel<0,128><<<gDV, blk, 0, stream>>>(T0, EMB16, nullptr, nullptr, (float*)d_out, nullptr, BN, BV, BD);
}
